// ScaledDotProduct_21586505630084
// MI455X (gfx1250) — matrix operand order repaired and checked by reading, not compiled, not run on hardware
//
#include <hip/hip_runtime.h>

typedef __attribute__((ext_vector_type(16))) _Float16 v16h;
typedef __attribute__((ext_vector_type(8)))  _Float16 v8h;
typedef __attribute__((ext_vector_type(4)))  _Float16 v4h;
typedef __attribute__((ext_vector_type(8)))  float    v8f;

#define S_LEN 2048
#define D_DIM 64
#define KBLK  64                   // keys per block iteration
#define NBLK  (S_LEN / KBLK)
#define WAVES 8
#define ROWS_PER_WG (WAVES * 16)   // 128 query rows per workgroup
#define NQBLK (S_LEN / ROWS_PER_WG)

union V16U { v16h v; v8h h[2]; };

#if __has_builtin(__builtin_amdgcn_permlane16)
#define HAVE_PERMLANE 1
__device__ __forceinline__ float permx16(float x, unsigned lo, unsigned hi) {
    return __uint_as_float(__builtin_amdgcn_permlane16(
        __float_as_uint(x), __float_as_uint(x), lo, hi, false, false));
}
#endif

// max-reduce across the 16 lanes of each half-wave (rows live per half-wave)
__device__ __forceinline__ float rowmax16(float v) {
#ifdef HAVE_PERMLANE
    v = fmaxf(v, permx16(v, 0x67452301u, 0xEFCDAB89u)); // lane ^= 1
    v = fmaxf(v, permx16(v, 0x54761032u, 0xDCFE98BAu)); // lane ^= 2
    v = fmaxf(v, permx16(v, 0x32107654u, 0xBA98FEDCu)); // lane ^= 4
    v = fmaxf(v, permx16(v, 0xFEDCBA98u, 0x76543210u)); // lane ^= 8
#else
    v = fmaxf(v, __shfl_xor(v, 1));
    v = fmaxf(v, __shfl_xor(v, 2));
    v = fmaxf(v, __shfl_xor(v, 4));
    v = fmaxf(v, __shfl_xor(v, 8));
#endif
    return v;
}

__global__ __launch_bounds__(256)
void fa_fwd_kernel(const float* __restrict__ Qg, const float* __restrict__ Kg,
                   const float* __restrict__ Vg, float* __restrict__ Og)
{
    __shared__ __align__(16) _Float16 Ksh[KBLK * D_DIM];          // [key][d]   8 KB
    __shared__ __align__(16) _Float16 Vt [D_DIM * KBLK];          // [d][key]   8 KB
    __shared__ __align__(16) _Float16 Plds[WAVES][16 * KBLK];     // [row][key] 16 KB

    const int tid  = threadIdx.x;
    const int wave = tid >> 5;
    const int lane = tid & 31;
    const int hi   = lane >> 4;    // 0: lanes 0-15, 1: lanes 16-31
    const int l16  = lane & 15;

    const int wg    = blockIdx.x;
    const int head  = wg / NQBLK;             // flattened (b,h): 0..63
    const int qblk  = wg % NQBLK;
    const size_t base  = (size_t)head * S_LEN * D_DIM;
    const int    qbase = qblk * ROWS_PER_WG + wave * 16;

    // ---- Q tile: two 16x32 f16 A-fragments (d=0..31, 32..63); scale folded ----
    const float qscale = 0.125f * 1.44269504088896340736f; // 1/sqrt(64) * log2(e)
    v16h qa[2];
    {
        const float* qrow = Qg + base + (size_t)(qbase + l16) * D_DIM;
        #pragma unroll
        for (int f = 0; f < 2; ++f) {
            const int d0 = f * 32 + hi * 16;
            const int d1 = d0 + 8;
            float4 x0 = *(const float4*)(qrow + d0);
            float4 x1 = *(const float4*)(qrow + d0 + 4);
            float4 x2 = *(const float4*)(qrow + d1);
            float4 x3 = *(const float4*)(qrow + d1 + 4);
            float t[16] = {x0.x,x0.y,x0.z,x0.w, x1.x,x1.y,x1.z,x1.w,
                           x2.x,x2.y,x2.z,x2.w, x3.x,x3.y,x3.z,x3.w};
            v16h a;
            #pragma unroll
            for (int i = 0; i < 16; ++i) a[i] = (_Float16)(t[i] * qscale);
            qa[f] = a;
        }
    }

    // constant all-ones B fragment: acc[4] = P @ ones == running row-sum l
    v16h onesB;
    #pragma unroll
    for (int i = 0; i < 16; ++i) onesB[i] = (_Float16)1.0f;

    // per-lane state: 8 rows (row = r + 8*hi); acc[0..3] = O chunks, acc[4] = l
    float m[8];
    v8f acc[5];
    #pragma unroll
    for (int r = 0; r < 8; ++r) m[r] = -1e30f;
    #pragma unroll
    for (int n = 0; n < 5; ++n) { v8f z = {}; acc[n] = z; }

    // staging coordinates
    const int krow = tid >> 2;            // 0..63
    const int kd0  = (tid & 3) * 16;      // 0,16,32,48
    const int vk4  = (tid & 15) * 4;      // key group of 4
    const int vd0  = (tid >> 4) * 4;      // 4 dims per thread
    const float* kptr = Kg + base + (size_t)krow * D_DIM + kd0;
    const float* vptr = Vg + base + (size_t)vk4  * D_DIM + vd0;

    for (int kb = 0; kb < NBLK; ++kb) {
        // ---- stage K (row-major f16) ----
        {
            float4 a0 = *(const float4*)(kptr);
            float4 a1 = *(const float4*)(kptr + 4);
            float4 a2 = *(const float4*)(kptr + 8);
            float4 a3 = *(const float4*)(kptr + 12);
            float t[16] = {a0.x,a0.y,a0.z,a0.w, a1.x,a1.y,a1.z,a1.w,
                           a2.x,a2.y,a2.z,a2.w, a3.x,a3.y,a3.z,a3.w};
            v8h h0, h1;
            #pragma unroll
            for (int i = 0; i < 8; ++i) { h0[i] = (_Float16)t[i]; h1[i] = (_Float16)t[8+i]; }
            *(v8h*)&Ksh[krow * D_DIM + kd0]     = h0;   // ds_store_b128
            *(v8h*)&Ksh[krow * D_DIM + kd0 + 8] = h1;   // ds_store_b128
        }
        // ---- stage V transposed: 4 keys x 4 dims per thread, packed b64 stores ----
        {
            float4 r0 = *(const float4*)(vptr);
            float4 r1 = *(const float4*)(vptr + D_DIM);
            float4 r2 = *(const float4*)(vptr + 2 * D_DIM);
            float4 r3 = *(const float4*)(vptr + 3 * D_DIM);
            float c0[4] = {r0.x, r1.x, r2.x, r3.x};
            float c1[4] = {r0.y, r1.y, r2.y, r3.y};
            float c2[4] = {r0.z, r1.z, r2.z, r3.z};
            float c3[4] = {r0.w, r1.w, r2.w, r3.w};
            float* cols[4] = {c0, c1, c2, c3};
            #pragma unroll
            for (int i = 0; i < 4; ++i) {
                v4h p;
                #pragma unroll
                for (int j = 0; j < 4; ++j) p[j] = (_Float16)cols[i][j];
                *(v4h*)&Vt[(vd0 + i) * KBLK + vk4] = p;  // ds_store_b64
            }
        }
        kptr += (size_t)KBLK * D_DIM;
        vptr += (size_t)KBLK * D_DIM;
        __syncthreads();

        // prefetch next block while this one computes (global_prefetch_b8, no cnt)
        if (kb + 1 < NBLK) {
            __builtin_prefetch(kptr, 0, 3);
            #pragma unroll
            for (int j = 0; j < 4; ++j) __builtin_prefetch(vptr + j * D_DIM, 0, 3);
        }

        // ---- S = Q @ K^T : 4 interleaved score tiles; tile t col N <-> key 4N+t ----
        v8f s[4];
        #pragma unroll
        for (int t = 0; t < 4; ++t) {
            const _Float16* kp = &Ksh[(4 * l16 + t) * D_DIM + hi * 16];
            V16U b0, b1;
            b0.h[0] = *(const v8h*)(kp);
            b0.h[1] = *(const v8h*)(kp + 8);
            b1.h[0] = *(const v8h*)(kp + 32);
            b1.h[1] = *(const v8h*)(kp + 40);
            v8f c = {};
            c = __builtin_amdgcn_wmma_f32_16x16x32_f16(false, qa[0], false, b0.v,
                                                       (short)0, c, false, false);
            c = __builtin_amdgcn_wmma_f32_16x16x32_f16(false, qa[1], false, b1.v,
                                                       (short)0, c, false, false);
            s[t] = c;
        }

        // ---- online softmax: max via permlane (VALU), sum via ones-WMMA ----
        float al[8];
        _Float16 ph[4][8];
        #pragma unroll
        for (int r = 0; r < 8; ++r) {
            float c = fmaxf(fmaxf(s[0][r], s[1][r]), fmaxf(s[2][r], s[3][r]));
            c = rowmax16(c);
            const float mn = fmaxf(m[r], c);
            al[r] = exp2f(m[r] - mn);
            m[r]  = mn;
            #pragma unroll
            for (int t = 0; t < 4; ++t)
                ph[t][r] = (_Float16)exp2f(s[t][r] - mn);
        }

        // ---- P relayout: packed b64 stores (4 consecutive keys per lane/row) ----
        _Float16* pw = &Plds[wave][0];
        #pragma unroll
        for (int r = 0; r < 8; ++r) {
            const int R = r + 8 * hi;
            v4h p = {ph[0][r], ph[1][r], ph[2][r], ph[3][r]};  // keys 4*l16 .. +3
            *(v4h*)&pw[R * KBLK + 4 * l16] = p;                 // ds_store_b64
        }
        asm volatile("" ::: "memory");   // same-wave DS ops are hardware-ordered
        V16U pa[2];
        const _Float16* pr = &Plds[wave][l16 * KBLK + hi * 8];
        pa[0].h[0] = *(const v8h*)(pr);
        pa[0].h[1] = *(const v8h*)(pr + 16);
        pa[1].h[0] = *(const v8h*)(pr + 32);
        pa[1].h[1] = *(const v8h*)(pr + 48);

        // ---- O += P @ V (4 chunks) and l += P @ 1 (chunk 4), alpha-rescaled ----
        #pragma unroll
        for (int kh = 0; kh < 2; ++kh) {
            #pragma unroll
            for (int n = 0; n < 5; ++n) {
                V16U b;
                if (n < 4) {
                    const _Float16* vb = &Vt[(n * 16 + l16) * KBLK + kh * 32 + hi * 8];
                    b.h[0] = *(const v8h*)(vb);
                    b.h[1] = *(const v8h*)(vb + 16);
                } else {
                    b.v = onesB;
                }
                v8f a = acc[n];
                if (kh == 0) {
                    #pragma unroll
                    for (int r = 0; r < 8; ++r) a[r] *= al[r];
                }
                acc[n] = __builtin_amdgcn_wmma_f32_16x16x32_f16(false, pa[kh].v, false, b.v,
                                                                (short)0, a, false, false);
            }
        }
        __syncthreads();   // protect Ksh/Vt before next staging
    }

    // ---- epilogue: O /= l, store f32 ----
    float linv[8];
    #pragma unroll
    for (int r = 0; r < 8; ++r) linv[r] = 1.0f / acc[4][r];
    float* ob = Og + base;
    #pragma unroll
    for (int n = 0; n < 4; ++n) {
        #pragma unroll
        for (int r = 0; r < 8; ++r) {
            const int R = r + 8 * hi;
            ob[(size_t)(qbase + R) * D_DIM + n * 16 + l16] = acc[n][r] * linv[r];
        }
    }
}

extern "C" void kernel_launch(void* const* d_in, const int* in_sizes, int n_in,
                              void* d_out, int out_size, void* d_ws, size_t ws_size,
                              hipStream_t stream) {
    const float* q = (const float*)d_in[0];
    const float* k = (const float*)d_in[1];
    const float* v = (const float*)d_in[2];
    float* o = (float*)d_out;
    const int n_heads = 4 * 16;                 // B*H
    dim3 grid(n_heads * NQBLK);                 // 64 * 16 = 1024 workgroups
    dim3 block(256);                            // 8 waves (wave32)
    fa_fwd_kernel<<<grid, block, 0, stream>>>(q, k, v, o);
}
